// Mamba2Block_6038724018222
// MI455X (gfx1250) — matrix operand order repaired and checked by reading, not compiled, not run on hardware
//
#include <hip/hip_runtime.h>

// ---------------- constants ----------------
enum : int {
  D_MODEL  = 2048,
  D_STATE  = 64,
  D_CONV   = 4,
  HEADDIM  = 64,
  NGROUPS  = 1,
  CHUNK    = 64,
  D_INNER  = 4096,
  NHEADS   = 64,
  CONV_DIM = D_INNER + 2 * NGROUPS * D_STATE,          // 4224
  D_IN_PROJ = 2 * D_INNER + 2 * NGROUPS * D_STATE + NHEADS, // 8384
  NPAD     = 8448,                                     // D_IN_PROJ padded to 128
  BATCH    = 2,
  SEQ      = 4096,
  NC       = SEQ / CHUNK                               // 64
};
#define EPSF 1e-5f

// ---------------- types ----------------
typedef __attribute__((ext_vector_type(16))) __bf16 v16bf;
typedef __attribute__((ext_vector_type(8)))  __bf16 v8bf;
typedef __attribute__((ext_vector_type(8)))  float  v8f;

union BF16x16 { v16bf v; v8bf h[2]; };

__device__ inline __bf16 f2bf(float f) {
  unsigned u = __builtin_bit_cast(unsigned, f);
  unsigned r = (u + 0x7fffu + ((u >> 16) & 1u)) >> 16;
  unsigned short s = (unsigned short)r;
  return __builtin_bit_cast(__bf16, s);
}
__device__ inline float bf2f(__bf16 b) {
  unsigned short s = __builtin_bit_cast(unsigned short, b);
  unsigned u = ((unsigned)s) << 16;
  return __builtin_bit_cast(float, u);
}
__device__ inline v8f wmma_bf16(v16bf a, v16bf b, v8f c) {
  return __builtin_amdgcn_wmma_f32_16x16x32_bf16(false, a, false, b, (short)0, c, false, false);
}
__device__ inline float siluf(float z) { return z / (1.f + __expf(-z)); }

// ---------------- conversion kernels ----------------
__global__ __launch_bounds__(256) void cvt_f32_bf16(const float* __restrict__ s,
                                                    __bf16* __restrict__ d, size_t n) {
  size_t i = (size_t)blockIdx.x * 256 + threadIdx.x;
  if (i < n) d[i] = f2bf(s[i]);
}
// convert + zero-pad rows beyond rows_valid (row-major [rows][cols])
__global__ __launch_bounds__(256) void cvt_pad_bf16(const float* __restrict__ s,
                                                    __bf16* __restrict__ d,
                                                    int rows_valid, int cols, size_t n) {
  size_t i = (size_t)blockIdx.x * 256 + threadIdx.x;
  if (i >= n) return;
  int row = (int)(i / cols);
  d[i] = (row < rows_valid) ? f2bf(s[i]) : f2bf(0.f);
}

// ---------------- bf16 WMMA GEMM:  C[M][ldc] = A[M][K] * B[Npad][K]^T ----------------
// block: 256 threads (8 waves), tile 128x128, K step 32
__global__ __launch_bounds__(256) void gemm_bf16_wmma(const __bf16* __restrict__ A,
                                                      const __bf16* __restrict__ B,
                                                      float* __restrict__ C,
                                                      int M, int Nvalid, int K, int ldc) {
  __shared__ __attribute__((aligned(16))) __bf16 As[128 * 32];
  __shared__ __attribute__((aligned(16))) __bf16 Bs[128 * 32];
  const int t = threadIdx.x;
  const int lane = t & 31, wave = t >> 5;
  const int wy = wave >> 2, wx = wave & 3;
  const int ml = lane & 15, kh = lane >> 4;
  const int bm = blockIdx.x * 128;
  const int bn = blockIdx.y * 128;

  v8f acc[4][2] = {};

  for (int k0 = 0; k0 < K; k0 += 32) {
    __syncthreads();
#pragma unroll
    for (int u = 0; u < 2; u++) {
      int chunk = t + u * 256;            // 0..511
      int row = chunk >> 2;               // 0..127
      int cg  = (chunk & 3) * 8;          // 0,8,16,24
      *(v8bf*)&As[row * 32 + cg] = *(const v8bf*)(A + (size_t)(bm + row) * K + k0 + cg);
      *(v8bf*)&Bs[row * 32 + cg] = *(const v8bf*)(B + (size_t)(bn + row) * K + k0 + cg);
    }
    if (k0 + 32 < K) {  // emits global_prefetch_b8
      __builtin_prefetch(A + (size_t)(bm + (t >> 1)) * K + k0 + 32, 0, 1);
      __builtin_prefetch(B + (size_t)(bn + (t >> 1)) * K + k0 + 32, 0, 1);
    }
    __syncthreads();

    BF16x16 afr[4], bfr[2];
#pragma unroll
    for (int i = 0; i < 4; i++) {
      const __bf16* ap = &As[(wy * 64 + i * 16 + ml) * 32];
      afr[i].h[0] = *(const v8bf*)(ap + kh * 8);
      afr[i].h[1] = *(const v8bf*)(ap + 16 + kh * 8);
    }
#pragma unroll
    for (int j = 0; j < 2; j++) {
      const __bf16* bp = &Bs[(wx * 32 + j * 16 + ml) * 32 + kh * 8];
      bfr[j].h[0] = *(const v8bf*)(bp);
      bfr[j].h[1] = *(const v8bf*)(bp + 16);
    }
#pragma unroll
    for (int i = 0; i < 4; i++)
#pragma unroll
      for (int j = 0; j < 2; j++)
        acc[i][j] = wmma_bf16(afr[i].v, bfr[j].v, acc[i][j]);
  }

#pragma unroll
  for (int i = 0; i < 4; i++)
#pragma unroll
    for (int j = 0; j < 2; j++)
#pragma unroll
      for (int v = 0; v < 8; v++) {
        int gm = bm + wy * 64 + i * 16 + v + kh * 8;
        int gn = bn + wx * 32 + j * 16 + ml;
        if (gn < Nvalid) C[(size_t)gm * ldc + gn] = acc[i][j][v];
      }
  (void)M;
}

// ---------------- dt softplus + per-chunk cumsum of dt*A ----------------
__global__ __launch_bounds__(128) void dt_kernel(const float* __restrict__ zx,
                                                 const float* __restrict__ dt_bias,
                                                 const float* __restrict__ A_log,
                                                 float* __restrict__ dt_soft,
                                                 float* __restrict__ Acum) {
  int idx = blockIdx.x * 128 + threadIdx.x;     // b*4096 + c*64 + h
  if (idx >= BATCH * NC * NHEADS) return;
  int h = idx & 63, c = (idx >> 6) & 63, b = idx >> 12;
  float Ah = -__expf(A_log[h]);
  float bias = dt_bias[h];
  float cum = 0.f;
  for (int s = 0; s < CHUNK; s++) {
    size_t tok = (size_t)b * SEQ + c * CHUNK + s;
    float raw = zx[tok * D_IN_PROJ + (D_INNER + CONV_DIM) + h] + bias;
    float dt = (raw > 20.f) ? raw : log1pf(__expf(raw));
    dt_soft[tok * NHEADS + h] = dt;
    cum += dt * Ah;
    Acum[(size_t)idx * CHUNK + s] = cum;
  }
}

// ---------------- depthwise conv4 + bias + SiLU + split/scale ----------------
__global__ __launch_bounds__(256) void conv_kernel(const float* __restrict__ zx,
                                                   const float* __restrict__ conv_w,
                                                   const float* __restrict__ conv_b,
                                                   const float* __restrict__ dt_soft,
                                                   __bf16* __restrict__ xs_bf,
                                                   __bf16* __restrict__ xdt_bf,
                                                   __bf16* __restrict__ B_bf,
                                                   __bf16* __restrict__ C_bf) {
  size_t idx = (size_t)blockIdx.x * 256 + threadIdx.x;
  const size_t total = (size_t)BATCH * SEQ * CONV_DIM;
  if (idx >= total) return;
  int ch = (int)(idx % CONV_DIM);
  size_t r = idx / CONV_DIM;
  int l = (int)(r % SEQ);
  int b = (int)(r / SEQ);
  float acc = conv_b[ch];
#pragma unroll
  for (int k = 0; k < D_CONV; k++) {
    int ll = l + k - (D_CONV - 1);
    if (ll >= 0)
      acc += conv_w[ch * D_CONV + k] * zx[((size_t)b * SEQ + ll) * D_IN_PROJ + D_INNER + ch];
  }
  float v = siluf(acc);
  size_t tok = (size_t)b * SEQ + l;
  if (ch < D_INNER) {
    xs_bf[tok * D_INNER + ch] = f2bf(v);
    int h = ch >> 6;
    xdt_bf[tok * D_INNER + ch] = f2bf(v * dt_soft[tok * NHEADS + h]);
  } else if (ch < D_INNER + D_STATE) {
    B_bf[tok * D_STATE + (ch - D_INNER)] = f2bf(v);
  } else {
    C_bf[tok * D_STATE + (ch - D_INNER - D_STATE)] = f2bf(v);
  }
}

// ---------------- SSD chunk kernel: Y_diag + chunk states (per b,c,h) ----------------
// block: 128 threads = 4 waves; one 64-token chunk of one head
__global__ __launch_bounds__(128) void ssd_chunk_kernel(const __bf16* __restrict__ Cg,
                                                        const __bf16* __restrict__ Bg,
                                                        const __bf16* __restrict__ Xdt,
                                                        const float* __restrict__ Acum,
                                                        float* __restrict__ Y,
                                                        float* __restrict__ states) {
  __shared__ __attribute__((aligned(16))) __bf16 Cs[64 * 64];
  __shared__ __attribute__((aligned(16))) __bf16 Bs[64 * 64];
  __shared__ __attribute__((aligned(16))) __bf16 Xt[64 * 64];  // [p][s]
  __shared__ __attribute__((aligned(16))) __bf16 Gs[64 * 64];  // masked (C B^T) [l][s]
  __shared__ float aa[64];
  __shared__ float dec[64];

  const int bid = blockIdx.x;                 // b*4096 + c*64 + h
  const int h = bid & 63, c = (bid >> 6) & 63, b = bid >> 12;
  const int t = threadIdx.x;
  const int lane = t & 31, wave = t >> 5;
  const int ml = lane & 15, kh = lane >> 4;
  const size_t lbase = (size_t)b * SEQ + c * CHUNK;

  // ---- stage tiles ----
  const int row = t >> 1, half = t & 1;       // 2 threads per row
  {
    const __bf16* cs = Cg + (lbase + row) * D_STATE + half * 32;
    const __bf16* bs = Bg + (lbase + row) * D_STATE + half * 32;
#pragma unroll
    for (int q = 0; q < 4; q++) {
      *(v8bf*)&Cs[row * 64 + half * 32 + q * 8] = *(const v8bf*)(cs + q * 8);
      *(v8bf*)&Bs[row * 64 + half * 32 + q * 8] = *(const v8bf*)(bs + q * 8);
    }
    const __bf16* xs = Xdt + (lbase + row) * D_INNER + h * HEADDIM + half * 32;
#pragma unroll
    for (int q = 0; q < 32; q++) Xt[(half * 32 + q) * 64 + row] = xs[q];  // transpose
  }
  if (t < 64) aa[t] = Acum[(size_t)bid * CHUNK + t];
  __syncthreads();
  if (t < 64) dec[t] = __expf(aa[63] - aa[t]);
  __syncthreads();

  // ---- phase 1: G = C * B^T  (contract n=D_STATE) ----
  v8f g[4] = {};
#pragma unroll
  for (int ks = 0; ks < 64; ks += 32) {
    BF16x16 af;
    const __bf16* ap = &Cs[(wave * 16 + ml) * 64 + ks];
    af.h[0] = *(const v8bf*)(ap + kh * 8);
    af.h[1] = *(const v8bf*)(ap + 16 + kh * 8);
#pragma unroll
    for (int j = 0; j < 4; j++) {
      BF16x16 bf_;
      const __bf16* bp = &Bs[(j * 16 + ml) * 64 + ks + kh * 8];
      bf_.h[0] = *(const v8bf*)(bp);
      bf_.h[1] = *(const v8bf*)(bp + 16);
      g[j] = wmma_bf16(af.v, bf_.v, g[j]);
    }
  }
  // ---- phase 2: mask with exp(segsum), to LDS bf16 ----
#pragma unroll
  for (int j = 0; j < 4; j++)
#pragma unroll
    for (int v = 0; v < 8; v++) {
      int lg = wave * 16 + v + kh * 8;
      int sg = j * 16 + ml;
      float val = (lg >= sg) ? g[j][v] * __expf(aa[lg] - aa[sg]) : 0.f;
      Gs[lg * 64 + sg] = f2bf(val);  // wave reads back only its own rows
    }

  // ---- phase 3: Y_diag = Gm * X  (contract s) ----
  v8f yd[4] = {};
#pragma unroll
  for (int ks = 0; ks < 64; ks += 32) {
    BF16x16 af;
    const __bf16* ap = &Gs[(wave * 16 + ml) * 64 + ks];
    af.h[0] = *(const v8bf*)(ap + kh * 8);
    af.h[1] = *(const v8bf*)(ap + 16 + kh * 8);
#pragma unroll
    for (int j = 0; j < 4; j++) {
      BF16x16 bf_;
      const __bf16* bp = &Xt[(j * 16 + ml) * 64 + ks + kh * 8];  // B[k=s][p]=x[s,p]=Xt[p][s]
      bf_.h[0] = *(const v8bf*)(bp);
      bf_.h[1] = *(const v8bf*)(bp + 16);
      yd[j] = wmma_bf16(af.v, bf_.v, yd[j]);
    }
  }
#pragma unroll
  for (int j = 0; j < 4; j++)
#pragma unroll
    for (int v = 0; v < 8; v++) {
      int lg = wave * 16 + v + kh * 8;
      int pg = j * 16 + ml;
      Y[(lbase + lg) * D_INNER + h * HEADDIM + pg] = yd[j][v];
    }

  // ---- phase 4: states[p][n] = sum_s x[s,p]*dec[s]*B[s,n] ----
  v8f st[4] = {};
#pragma unroll
  for (int ks = 0; ks < 64; ks += 32) {
    BF16x16 af;
    int p = wave * 16 + ml;
#pragma unroll
    for (int e = 0; e < 8; e++) {
      int s1 = ks + kh * 8 + e;
      int s2 = ks + 16 + kh * 8 + e;
      af.h[0][e] = f2bf(bf2f(Xt[p * 64 + s1]) * dec[s1]);
      af.h[1][e] = f2bf(bf2f(Xt[p * 64 + s2]) * dec[s2]);
    }
#pragma unroll
    for (int j = 0; j < 4; j++) {
      BF16x16 bf_;
#pragma unroll
      for (int e = 0; e < 16; e++)  // B[k=s][n]: gather Bs[s][n]
        bf_.v[e] = Bs[(ks + kh * 8 + (e & 7) + (e >> 3) * 16) * 64 + j * 16 + ml];
      st[j] = wmma_bf16(af.v, bf_.v, st[j]);
    }
  }
#pragma unroll
  for (int j = 0; j < 4; j++)
#pragma unroll
    for (int v = 0; v < 8; v++) {
      int pg = wave * 16 + v + kh * 8;
      int ng = j * 16 + ml;
      states[(size_t)bid * (HEADDIM * D_STATE) + pg * D_STATE + ng] = st[j][v];
    }
}

// ---------------- inter-chunk scan: states -> exclusive-prefix prev_states (in place) ----
__global__ __launch_bounds__(256) void scan_kernel(const float* __restrict__ Acum,
                                                   float* __restrict__ states) {
  const int bh = blockIdx.x;                // b*64 + h
  const int b = bh >> 6, h = bh & 63;
  const int t = threadIdx.x;
  float hreg[16];
#pragma unroll
  for (int i = 0; i < 16; i++) hreg[i] = 0.f;
  for (int c = 0; c < NC; c++) {
    size_t idx = ((size_t)b * NC + c) * NHEADS + h;     // == bid of chunk kernel
    float dcy = __expf(Acum[idx * CHUNK + CHUNK - 1]);
    float* sp = states + idx * (HEADDIM * D_STATE);
#pragma unroll
    for (int i = 0; i < 16; i++) {
      float s = sp[t + i * 256];
      sp[t + i * 256] = hreg[i];            // write prev (exclusive prefix)
      hreg[i] = dcy * hreg[i] + s;
    }
  }
}

// ---------------- Y_off = exp(a_l) * C * prev^T, combine + gate ----------------
__global__ __launch_bounds__(128) void yoff_kernel(const __bf16* __restrict__ Cg,
                                                   const float* __restrict__ prev,
                                                   const float* __restrict__ Acum,
                                                   const float* __restrict__ zx,
                                                   const __bf16* __restrict__ xs_bf,
                                                   const float* __restrict__ Dv,
                                                   float* __restrict__ Y) {
  __shared__ __attribute__((aligned(16))) __bf16 Cs[64 * 64];
  __shared__ __attribute__((aligned(16))) __bf16 Ps[64 * 64];  // prev [p][n]
  __shared__ float aa[64];

  const int bid = blockIdx.x;
  const int h = bid & 63, c = (bid >> 6) & 63, b = bid >> 12;
  const int t = threadIdx.x;
  const int lane = t & 31, wave = t >> 5;
  const int ml = lane & 15, kh = lane >> 4;
  const size_t lbase = (size_t)b * SEQ + c * CHUNK;

  const int row = t >> 1, half = t & 1;
  {
    const __bf16* cs = Cg + (lbase + row) * D_STATE + half * 32;
#pragma unroll
    for (int q = 0; q < 4; q++)
      *(v8bf*)&Cs[row * 64 + half * 32 + q * 8] = *(const v8bf*)(cs + q * 8);
    const float* sp = prev + (size_t)bid * (HEADDIM * D_STATE) + row * 64 + half * 32;
#pragma unroll
    for (int q = 0; q < 32; q++) Ps[row * 64 + half * 32 + q] = f2bf(sp[q]);
  }
  if (t < 64) aa[t] = Acum[(size_t)bid * CHUNK + t];
  __syncthreads();

  v8f f[4] = {};
#pragma unroll
  for (int ks = 0; ks < 64; ks += 32) {     // contract n=D_STATE
    BF16x16 af;
    const __bf16* ap = &Cs[(wave * 16 + ml) * 64 + ks];
    af.h[0] = *(const v8bf*)(ap + kh * 8);
    af.h[1] = *(const v8bf*)(ap + 16 + kh * 8);
#pragma unroll
    for (int j = 0; j < 4; j++) {
      BF16x16 bf_;
      const __bf16* bp = &Ps[(j * 16 + ml) * 64 + ks + kh * 8];  // B[k=n][p]=prev[p][n]
      bf_.h[0] = *(const v8bf*)(bp);
      bf_.h[1] = *(const v8bf*)(bp + 16);
      f[j] = wmma_bf16(af.v, bf_.v, f[j]);
    }
  }

  const float Dh = Dv[h];
#pragma unroll
  for (int j = 0; j < 4; j++)
#pragma unroll
    for (int v = 0; v < 8; v++) {
      int lg = wave * 16 + v + kh * 8;
      int pg = j * 16 + ml;
      size_t tok = lbase + lg;
      int d = h * HEADDIM + pg;
      float yoff = f[j][v] * __expf(aa[lg]);
      float xv = bf2f(xs_bf[tok * D_INNER + d]);
      float zv = zx[tok * D_IN_PROJ + d];                 // z = first D_INNER cols
      float y = Y[tok * D_INNER + d] + yoff + Dh * xv;
      Y[tok * D_INNER + d] = y * siluf(zv);
    }
}

// ---------------- RMSNorm (per token over D_INNER) -> bf16 ----------------
__global__ __launch_bounds__(256) void rmsnorm_kernel(const float* __restrict__ Y,
                                                      const float* __restrict__ norm_w,
                                                      __bf16* __restrict__ ybf) {
  __shared__ float red[256];
  const size_t tok = blockIdx.x;
  const int t = threadIdx.x;
  const float* yp = Y + tok * D_INNER;
  float ss = 0.f;
  for (int i = t; i < D_INNER; i += 256) { float v = yp[i]; ss += v * v; }
  red[t] = ss;
  __syncthreads();
  for (int s = 128; s > 0; s >>= 1) {
    if (t < s) red[t] += red[t + s];
    __syncthreads();
  }
  float scale = rsqrtf(red[0] / (float)D_INNER + EPSF);
  for (int i = t; i < D_INNER; i += 256)
    ybf[tok * D_INNER + i] = f2bf(yp[i] * scale * norm_w[i]);
}

// ---------------- host ----------------
static inline size_t alignup(size_t x) { return (x + 255) & ~(size_t)255; }

extern "C" void kernel_launch(void* const* d_in, const int* in_sizes, int n_in,
                              void* d_out, int out_size, void* d_ws, size_t ws_size,
                              hipStream_t stream) {
  const float* x          = (const float*)d_in[0];
  const float* in_proj_w  = (const float*)d_in[1];
  const float* conv_w     = (const float*)d_in[2];
  const float* conv_b     = (const float*)d_in[3];
  const float* dt_bias    = (const float*)d_in[4];
  const float* A_log      = (const float*)d_in[5];
  const float* Dv         = (const float*)d_in[6];
  const float* norm_w     = (const float*)d_in[7];
  const float* out_proj_w = (const float*)d_in[8];
  float* out = (float*)d_out;
  (void)in_sizes; (void)n_in; (void)out_size; (void)ws_size;

  char* base = (char*)d_ws;
  size_t off = 0;
  auto carve = [&](size_t bytes) -> void* {
    void* r = base + off;
    off = alignup(off + bytes);
    return r;
  };
  const size_t NTOK = (size_t)BATCH * SEQ;                  // 8192
  __bf16* Xbf    = (__bf16*)carve(NTOK * D_MODEL * 2);
  __bf16* Wbf    = (__bf16*)carve((size_t)NPAD * D_MODEL * 2);
  __bf16* OWbf   = (__bf16*)carve((size_t)D_MODEL * D_INNER * 2);
  float*  ZX     = (float*) carve(NTOK * D_IN_PROJ * 4);
  float*  DTs    = (float*) carve(NTOK * NHEADS * 4);
  float*  ACUM   = (float*) carve((size_t)BATCH * NC * NHEADS * CHUNK * 4);
  __bf16* XSbf   = (__bf16*)carve(NTOK * D_INNER * 2);
  __bf16* XDTbf  = (__bf16*)carve(NTOK * D_INNER * 2);
  __bf16* Bbf    = (__bf16*)carve(NTOK * D_STATE * 2);
  __bf16* Cbf    = (__bf16*)carve(NTOK * D_STATE * 2);
  float*  STATES = (float*) carve((size_t)BATCH * NC * NHEADS * HEADDIM * D_STATE * 4);
  float*  Ybuf   = (float*) carve(NTOK * D_INNER * 4);
  __bf16* YBF    = (__bf16*)carve(NTOK * D_INNER * 2);

  // 1) convert to bf16
  {
    size_t n = NTOK * D_MODEL;
    cvt_f32_bf16<<<(unsigned)((n + 255) / 256), 256, 0, stream>>>(x, Xbf, n);
  }
  {
    size_t n = (size_t)NPAD * D_MODEL;
    cvt_pad_bf16<<<(unsigned)((n + 255) / 256), 256, 0, stream>>>(in_proj_w, Wbf,
                                                                  D_IN_PROJ, D_MODEL, n);
  }
  {
    size_t n = (size_t)D_MODEL * D_INNER;
    cvt_f32_bf16<<<(unsigned)((n + 255) / 256), 256, 0, stream>>>(out_proj_w, OWbf, n);
  }

  // 2) in_proj GEMM: ZX[8192][8384] = Xbf * Wbf^T
  {
    dim3 grid(NTOK / 128, NPAD / 128);
    gemm_bf16_wmma<<<grid, 256, 0, stream>>>(Xbf, Wbf, ZX,
                                             (int)NTOK, D_IN_PROJ, D_MODEL, D_IN_PROJ);
  }

  // 3) dt softplus + dA cumsum
  dt_kernel<<<(BATCH * NC * NHEADS + 127) / 128, 128, 0, stream>>>(ZX, dt_bias, A_log,
                                                                   DTs, ACUM);

  // 4) depthwise conv + SiLU + split
  {
    size_t n = NTOK * CONV_DIM;
    conv_kernel<<<(unsigned)((n + 255) / 256), 256, 0, stream>>>(ZX, conv_w, conv_b, DTs,
                                                                 XSbf, XDTbf, Bbf, Cbf);
  }

  // 5) SSD intra-chunk (Y_diag + states)
  ssd_chunk_kernel<<<BATCH * NC * NHEADS, 128, 0, stream>>>(Cbf, Bbf, XDTbf, ACUM,
                                                            Ybuf, STATES);

  // 6) inter-chunk scan (states -> prev_states, in place)
  scan_kernel<<<BATCH * NHEADS, 256, 0, stream>>>(ACUM, STATES);

  // 7) Y_off + D*x + SiLU(z) gating
  yoff_kernel<<<BATCH * NC * NHEADS, 128, 0, stream>>>(Cbf, STATES, ACUM, ZX, XSbf, Dv,
                                                       Ybuf);

  // 8) RMSNorm -> bf16
  rmsnorm_kernel<<<(unsigned)NTOK, 256, 0, stream>>>(Ybuf, norm_w, YBF);

  // 9) out_proj GEMM: out[8192][2048] = YBF * OWbf^T
  {
    dim3 grid(NTOK / 128, D_MODEL / 128);
    gemm_bf16_wmma<<<grid, 256, 0, stream>>>(YBF, OWbf, out,
                                             (int)NTOK, D_MODEL, D_INNER, D_MODEL);
  }
}
